// ATTENTION_79645873537440
// MI455X (gfx1250) — matrix operand order repaired and checked by reading, not compiled, not run on hardware
//
#include <hip/hip_runtime.h>

typedef __attribute__((ext_vector_type(16))) _Float16 v16h;
typedef __attribute__((ext_vector_type(8)))  float    v8f;

#define WMMA_F16(a, b, c) \
  __builtin_amdgcn_wmma_f32_16x16x32_f16(false, (a), false, (b), (short)0, (c), false, false)

// ---- fragment loaders (contiguous k order, k = 16*(lane/16) + s, wave32) -------------------

// A-matrix 16x32 f16 from LDS: lane m=l&15; lanes0-15 K {0..15}, lanes16-31 K {16..31}
__device__ __forceinline__ v16h ldsA(const _Float16* base, int ld, int row0, int k0, int lane) {
  int m  = lane & 15;
  int ko = (lane < 16) ? 0 : 16;
  const _Float16* p = base + (row0 + m) * ld + k0;
  v16h a;
#pragma unroll
  for (int i = 0; i < 8; ++i) a[i]     = p[ko + i];        // 8 contiguous halves -> b128
#pragma unroll
  for (int i = 0; i < 8; ++i) a[8 + i] = p[ko + 8 + i];   // 8 contiguous halves -> b128
  return a;
}

// B-matrix 32x16 f16 where B[k][n] = mat[n][k] (mat row-major, k contiguous, f16 in LDS)
__device__ __forceinline__ v16h ldsBt(const _Float16* base, int ld, int n0, int k0, int lane) {
  int n  = lane & 15;
  int ko = (lane < 16) ? 0 : 16;
  const _Float16* p = base + (n0 + n) * ld + k0 + ko;
  v16h b;
#pragma unroll
  for (int i = 0; i < 16; ++i) b[i] = p[i];
  return b;
}

// Same but mat is pre-converted f16 in global memory
__device__ __forceinline__ v16h glbBh(const _Float16* base, int ld, int n0, int k0, int lane) {
  int n  = lane & 15;
  int ko = (lane < 16) ? 0 : 16;
  const _Float16* p = base + (n0 + n) * ld + k0 + ko;
  v16h b;
#pragma unroll
  for (int i = 0; i < 16; ++i) b[i] = p[i];
  return b;
}

// conv2 implicit-GEMM A fragment: K index = tap*64 + cin, built from f32 h in LDS
__device__ __forceinline__ v16h convA(const float* h, int mt, int dd, int cin0, int lane) {
  int m  = lane & 15;
  int ko = (lane < 16) ? 0 : 16;
  int t  = mt * 16 + m;
  int r  = t >> 6, j = t & 63;
  int di = dd / 3, dj = dd % 3;
  int ri = r + di - 1, jj = j + dj - 1;
  bool valid = (ri >= 0 && ri < 3 && jj >= 0 && jj < 64);
  v16h a;
  if (valid) {
    const float* p = h + ((ri * 64 + jj) * 64 + cin0);
#pragma unroll
    for (int i = 0; i < 8; ++i) a[i]     = (_Float16)p[ko + i];
#pragma unroll
    for (int i = 0; i < 8; ++i) a[8 + i] = (_Float16)p[ko + 8 + i];
  } else {
#pragma unroll
    for (int i = 0; i < 16; ++i) a[i] = (_Float16)0.0f;
  }
  return a;
}

// ---------------------------------------------------------------------------
// Kernel 0: one-shot f32 -> f16 weight conversion into workspace
// ---------------------------------------------------------------------------
__global__ __launch_bounds__(256) void cvt_weights(
    const float* __restrict__ w2, const float* __restrict__ w3,
    const float* __restrict__ wq, const float* __restrict__ wk,
    const float* __restrict__ wv, const float* __restrict__ wp,
    _Float16* __restrict__ w2h, _Float16* __restrict__ w3h,
    _Float16* __restrict__ wqh, _Float16* __restrict__ wkh,
    _Float16* __restrict__ wvh, _Float16* __restrict__ wph)
{
  int i = blockIdx.x * 256 + threadIdx.x;
  if (i < 768 * 1536) w2h[i] = (_Float16)w2[i];
  if (i < 64 * 768)   w3h[i] = (_Float16)w3[i];
  if (i < 64 * 64) {
    wqh[i] = (_Float16)wq[i];
    wkh[i] = (_Float16)wk[i];
    wvh[i] = (_Float16)wv[i];
    wph[i] = (_Float16)wp[i];
  }
}

// ---------------------------------------------------------------------------
// Phase 1: per-sample  dense -> conv1 -> inorm -> attention -> conv2 -> ws
// grid = 4096 blocks, 256 threads (8 waves); one sample resident in LDS.
// ---------------------------------------------------------------------------
__global__ __launch_bounds__(256) void attn_phase1(
    const float* __restrict__ x,    const float* __restrict__ w1,  const float* __restrict__ b1,
    const float* __restrict__ chw,  const float* __restrict__ chb,
    const float* __restrict__ gnw,  const float* __restrict__ gnb,
    const _Float16* __restrict__ wqh, const float* __restrict__ bq,
    const _Float16* __restrict__ wkh, const float* __restrict__ bk,
    const _Float16* __restrict__ wvh, const float* __restrict__ bv,
    const _Float16* __restrict__ wph, const float* __restrict__ bp,
    const float* __restrict__ c2w_g, const float* __restrict__ c2b_g,
    _Float16* __restrict__ aout)
{
  extern __shared__ char smem[];
  float*    h   = (float*)   (smem + 0);       // [192][64] f32 (residual stream)
  _Float16* hn  = (_Float16*)(smem + 49152);   // [192][64] f16 (normed; later attn-out O)
  _Float16* qm  = (_Float16*)(smem + 73728);   // [192][64] f16
  _Float16* km  = (_Float16*)(smem + 98304);   // [192][64] f16
  _Float16* vT  = (_Float16*)(smem + 122880);  // [64][192] f16 (V transposed)
  float*    sc  = (float*)   (smem + 147456);  // [192][192] f32 scores; probs f16 in-place; later conv2 wts
  float*    h0  = (float*)   (smem + 294912);  // [3][64]
  float*    nsc = (float*)   (smem + 295680);  // [64] norm scale
  float*    nsh = (float*)   (smem + 295936);  // [64] norm shift
  float*    rcp = (float*)   (smem + 296192);  // [192] 1/rowsum
  float*    cw  = (float*)   (smem + 296960);  // [576] conv1 w
  float*    cb  = (float*)   (smem + 299264);  // [64]  conv1 b

  const int tid  = threadIdx.x;
  const int lane = tid & 31;
  const int wv8  = tid >> 5;
  const int s    = blockIdx.x;

  // ---- stage conv1 weights; dense(1->64) ----
  for (int i = tid; i < 576; i += 256) cw[i] = chw[i];
  if (tid < 64) cb[tid] = chb[tid];
  if (tid < 192) h0[tid] = x[s * 3 + (tid >> 6)] * w1[tid & 63] + b1[tid & 63];
  __syncthreads();

  // ---- conv1: (1,3,64) -> (64,3,64) ; token t = r*64 + j ----
  if (tid < 192) {
    int r = tid >> 6, j = tid & 63;
    float nb[3][3];
#pragma unroll
    for (int di = 0; di < 3; ++di)
#pragma unroll
      for (int dj = 0; dj < 3; ++dj) {
        int ri = r + di - 1, jj = j + dj - 1;
        nb[di][dj] = (ri >= 0 && ri < 3 && jj >= 0 && jj < 64) ? h0[ri * 64 + jj] : 0.0f;
      }
    for (int c = 0; c < 64; ++c) {
      float acc = cb[c];
#pragma unroll
      for (int di = 0; di < 3; ++di)
#pragma unroll
        for (int dj = 0; dj < 3; ++dj)
          acc += nb[di][dj] * cw[c * 9 + di * 3 + dj];
      h[tid * 64 + c] = acc;
    }
  }
  __syncthreads();

  // ---- instance norm stats per channel ----
  if (tid < 64) {
    float sum = 0.f, sq = 0.f;
    for (int t = 0; t < 192; ++t) { float v = h[t * 64 + tid]; sum += v; sq += v * v; }
    float mu = sum * (1.0f / 192.0f);
    float var = sq * (1.0f / 192.0f) - mu * mu;
    float rs = rsqrtf(var + 1e-5f) * gnw[tid];
    nsc[tid] = rs;
    nsh[tid] = gnb[tid] - mu * rs;
  }
  __syncthreads();
  if (tid < 192)
    for (int c = 0; c < 64; ++c)
      hn[tid * 64 + c] = (_Float16)(h[tid * 64 + c] * nsc[c] + nsh[c]);
  __syncthreads();

  const int nn   = lane & 15;
  const int moff = (lane < 16) ? 0 : 8;

  // ---- Q,K,V projections: 3 * (192x64 @ 64x64), 144 tile jobs over 8 waves ----
  for (int job = wv8; job < 144; job += 8) {
    int p = job / 48, t = job % 48;
    int mt = t >> 2, nt = t & 3;
    const _Float16* W  = (p == 0) ? wqh : ((p == 1) ? wkh : wvh);
    const float*    bb = (p == 0) ? bq  : ((p == 1) ? bk  : bv);
    v8f acc = {};
#pragma unroll
    for (int ks = 0; ks < 2; ++ks) {
      v16h a = ldsA(hn, 64, mt * 16, ks * 32, lane);
      v16h b = glbBh(W, 64, nt * 16, ks * 32, lane);
      acc = WMMA_F16(a, b, acc);
    }
    float bias = bb[nt * 16 + nn];
    if (p < 2) {
      _Float16* dst = (p == 0) ? qm : km;
#pragma unroll
      for (int i = 0; i < 8; ++i)
        dst[(mt * 16 + moff + i) * 64 + nt * 16 + nn] = (_Float16)(acc[i] + bias);
    } else {
#pragma unroll
      for (int i = 0; i < 8; ++i)   // store V transposed: vT[c][t], t contiguous
        vT[(nt * 16 + nn) * 192 + mt * 16 + moff + i] = (_Float16)(acc[i] + bias);
    }
  }
  __syncthreads();

  // ---- scores = Q @ K^T * 0.125 : 12x12 tiles ----
  for (int job = wv8; job < 144; job += 8) {
    int mt = job / 12, nt = job % 12;
    v8f acc = {};
#pragma unroll
    for (int ks = 0; ks < 2; ++ks) {
      v16h a = ldsA(qm, 64, mt * 16, ks * 32, lane);
      v16h b = ldsBt(km, 64, nt * 16, ks * 32, lane);
      acc = WMMA_F16(a, b, acc);
    }
#pragma unroll
    for (int i = 0; i < 8; ++i)
      sc[(mt * 16 + moff + i) * 192 + nt * 16 + nn] = acc[i] * 0.125f;
  }
  __syncthreads();

  // ---- softmax: unnormalized exp written f16 in place; keep 1/sum ----
  if (tid < 192) {
    float* row = sc + tid * 192;
    _Float16* rowh = (_Float16*)row;   // f16 row overlays first half of f32 row; safe in-place
    float mx = -3.0e38f;
    for (int i = 0; i < 192; ++i) mx = fmaxf(mx, row[i]);
    float ssum = 0.f;
    for (int i = 0; i < 192; ++i) {
      float e = __expf(row[i] - mx);
      ssum += e;
      rowh[i] = (_Float16)e;
    }
    rcp[tid] = 1.0f / ssum;
  }
  __syncthreads();

  // ---- O = P @ V (then scale by 1/rowsum): 12x4 tiles, K=192 ----
  for (int job = wv8; job < 48; job += 8) {
    int mt = job >> 2, nt = job & 3;
    v8f acc = {};
#pragma unroll
    for (int ks = 0; ks < 6; ++ks) {
      v16h a = ldsA((const _Float16*)sc, 384, mt * 16, ks * 32, lane); // probs f16, row stride 384 halves
      v16h b = ldsBt(vT, 192, nt * 16, ks * 32, lane);
      acc = WMMA_F16(a, b, acc);
    }
#pragma unroll
    for (int i = 0; i < 8; ++i) {
      int t = mt * 16 + moff + i;
      hn[t * 64 + nt * 16 + nn] = (_Float16)(acc[i] * rcp[t]);  // reuse hn as O
    }
  }
  __syncthreads();

  // ---- stage reordered conv2 weights into sc (free now): [16][tap*64+cin], rows 8..15 zero ----
  _Float16* w2r = (_Float16*)sc;
  for (int i = tid; i < 9216; i += 256) {
    int c = i / 576, kk = i % 576;
    int dd = kk >> 6, cin = kk & 63;
    w2r[i] = (c < 8) ? (_Float16)c2w_g[c * 576 + cin * 9 + dd] : (_Float16)0.0f;
  }

  // ---- output projection + residual into h ----
  for (int job = wv8; job < 48; job += 8) {
    int mt = job >> 2, nt = job & 3;
    v8f acc = {};
#pragma unroll
    for (int ks = 0; ks < 2; ++ks) {
      v16h a = ldsA(hn, 64, mt * 16, ks * 32, lane);
      v16h b = glbBh(wph, 64, nt * 16, ks * 32, lane);
      acc = WMMA_F16(a, b, acc);
    }
    float bias = bp[nt * 16 + nn];
#pragma unroll
    for (int i = 0; i < 8; ++i) {
      int idx = (mt * 16 + moff + i) * 64 + nt * 16 + nn;
      h[idx] += acc[i] + bias;
    }
  }
  __syncthreads();

  // ---- conv2 as implicit GEMM: out(192,16pad) = im2col(192,576) @ w2r^T ----
  for (int job = wv8; job < 12; job += 8) {
    int mt = job;
    v8f acc = {};
    for (int ks = 0; ks < 18; ++ks) {
      int dd = ks >> 1, cin0 = (ks & 1) * 32;
      v16h a = convA(h, mt, dd, cin0, lane);
      v16h b = ldsBt(w2r, 576, 0, ks * 32, lane);
      acc = WMMA_F16(a, b, acc);
    }
    if (nn < 8) {
      float bias = c2b_g[nn];
#pragma unroll
      for (int i = 0; i < 8; ++i)   // flatten index = c*192 + t, t contiguous -> vector store
        aout[(size_t)s * 1536 + nn * 192 + mt * 16 + moff + i] = (_Float16)(acc[i] + bias);
    }
  }
}

// ---------------------------------------------------------------------------
// Phase 2: batched FC chain. 128 blocks x 32 rows. WMMA for FC2/FC3.
// ---------------------------------------------------------------------------
__global__ __launch_bounds__(256) void attn_phase2(
    const _Float16* __restrict__ aout,
    const _Float16* __restrict__ w2h, const float* __restrict__ b2,
    const _Float16* __restrict__ w3h, const float* __restrict__ b3,
    const float* __restrict__ w4, const float* __restrict__ b4,
    float* __restrict__ out)
{
  extern __shared__ char smem[];
  _Float16* aL = (_Float16*)(smem + 0);        // [32][1536]
  _Float16* h2 = (_Float16*)(smem + 98304);    // [32][768]
  _Float16* h3 = (_Float16*)(smem + 147456);   // [32][64]

  const int tid  = threadIdx.x;
  const int lane = tid & 31;
  const int wv8  = tid >> 5;
  const int m0   = blockIdx.x * 32;
  const int nn   = lane & 15;
  const int moff = (lane < 16) ? 0 : 8;

  // stage 32x1536 f16 activations
  {
    const uint4* src = (const uint4*)(aout + (size_t)m0 * 1536);
    uint4* dst = (uint4*)aL;
    for (int i = tid; i < 6144; i += 256) dst[i] = src[i];
  }
  __syncthreads();

  // FC2: 32x768, K=1536 -> 2x48 tiles over 8 waves, relu
  for (int job = wv8; job < 96; job += 8) {
    int mt = job / 48, nt = job % 48;
    v8f acc = {};
    for (int ks = 0; ks < 48; ++ks) {
      v16h a = ldsA(aL, 1536, mt * 16, ks * 32, lane);
      v16h b = glbBh(w2h, 1536, nt * 16, ks * 32, lane);
      acc = WMMA_F16(a, b, acc);
    }
    float bias = b2[nt * 16 + nn];
#pragma unroll
    for (int i = 0; i < 8; ++i)
      h2[(mt * 16 + moff + i) * 768 + nt * 16 + nn] = (_Float16)fmaxf(acc[i] + bias, 0.0f);
  }
  __syncthreads();

  // FC3: 32x64, K=768 -> 2x4 tiles (one per wave), relu
  for (int job = wv8; job < 8; job += 8) {
    int mt = job >> 2, nt = job & 3;
    v8f acc = {};
    for (int ks = 0; ks < 24; ++ks) {
      v16h a = ldsA(h2, 768, mt * 16, ks * 32, lane);
      v16h b = glbBh(w3h, 768, nt * 16, ks * 32, lane);
      acc = WMMA_F16(a, b, acc);
    }
    float bias = b3[nt * 16 + nn];
#pragma unroll
    for (int i = 0; i < 8; ++i)
      h3[(mt * 16 + moff + i) * 64 + nt * 16 + nn] = (_Float16)fmaxf(acc[i] + bias, 0.0f);
  }
  __syncthreads();

  // FC4: 64-MAC dot per row
  if (tid < 32) {
    float acc = b4[0];
    for (int c = 0; c < 64; ++c) acc += (float)h3[tid * 64 + c] * w4[c];
    out[m0 + tid] = acc;
  }
}

extern "C" void kernel_launch(void* const* d_in, const int* in_sizes, int n_in,
                              void* d_out, int out_size, void* d_ws, size_t ws_size,
                              hipStream_t stream) {
  const float* x    = (const float*)d_in[0];
  const float* w1   = (const float*)d_in[1];
  const float* b1   = (const float*)d_in[2];
  const float* chw  = (const float*)d_in[3];
  const float* chb  = (const float*)d_in[4];
  const float* gnw  = (const float*)d_in[5];
  const float* gnb  = (const float*)d_in[6];
  const float* wq   = (const float*)d_in[7];
  const float* bq   = (const float*)d_in[8];
  const float* wk   = (const float*)d_in[9];
  const float* bk   = (const float*)d_in[10];
  const float* wv   = (const float*)d_in[11];
  const float* bv   = (const float*)d_in[12];
  const float* wp   = (const float*)d_in[13];
  const float* bp   = (const float*)d_in[14];
  const float* c2w  = (const float*)d_in[15];
  const float* c2b  = (const float*)d_in[16];
  const float* w2   = (const float*)d_in[17];
  const float* b2   = (const float*)d_in[18];
  const float* w3   = (const float*)d_in[19];
  const float* b3   = (const float*)d_in[20];
  const float* w4   = (const float*)d_in[21];
  const float* b4   = (const float*)d_in[22];
  float* out = (float*)d_out;

  // workspace layout (halves)
  _Float16* ws16 = (_Float16*)d_ws;
  _Float16* aFl  = ws16;                       // 4096*1536
  _Float16* w2h  = ws16 + 6291456;             // 768*1536
  _Float16* w3h  = w2h  + 1179648;             // 64*768
  _Float16* wqh  = w3h  + 49152;               // 64*64
  _Float16* wkh  = wqh  + 4096;
  _Float16* wvh  = wkh  + 4096;
  _Float16* wph  = wvh  + 4096;                // total ~15.1 MB

  cvt_weights<<<4608, 256, 0, stream>>>(w2, w3, wq, wk, wv, wp,
                                        w2h, w3h, wqh, wkh, wvh, wph);
  attn_phase1<<<4096, 256, 299520, stream>>>(
      x, w1, b1, chw, chb, gnw, gnb, wqh, bq, wkh, bk, wvh, bv, wph, bp,
      c2w, c2b, aFl);
  attn_phase2<<<128, 256, 151552, stream>>>(aFl, w2h, b2, w3h, b3, w4, b4, out);
}
